// Generator_85676007620777
// MI455X (gfx1250) — compile-verified, matrix operand order checked by reading, not run on hardware
//
#include <hip/hip_runtime.h>
#include <hip/hip_bf16.h>
#include <math.h>

// ---------- types ----------
typedef __attribute__((ext_vector_type(16))) __bf16 v16bf;
typedef __attribute__((ext_vector_type(8)))  float  v8f;

union Frag128 { uint4 q[2]; v16bf b; };

// ---------- optional CDNA5 async global->LDS path ----------
#if defined(__has_builtin)
#if __has_builtin(__builtin_amdgcn_global_load_async_to_lds_b128)
#define USE_ASYNC_LDS 1
typedef int v4i_t __attribute__((__vector_size__(4 * sizeof(int))));
typedef __attribute__((address_space(1))) v4i_t* as1_v4i;
typedef __attribute__((address_space(3))) v4i_t* as3_v4i;
#endif
#if __has_builtin(__builtin_amdgcn_s_wait_asynccnt)
#define HAVE_WAIT_ASYNC 1
#endif
#endif

__device__ __forceinline__ void wait_async0() {
#if defined(USE_ASYNC_LDS)
#if defined(HAVE_WAIT_ASYNC)
  __builtin_amdgcn_s_wait_asynccnt(0);
#else
  asm volatile("s_wait_asynccnt 0x0" ::: "memory");
#endif
#endif
}

__device__ __forceinline__ unsigned short f2bf(float f) {
  unsigned int u = __float_as_uint(f);
  unsigned int r = u + 0x7FFFu + ((u >> 16) & 1u);   // round-to-nearest-even
  if ((u & 0x7F800000u) == 0x7F800000u) r = u;       // inf/nan: truncate
  return (unsigned short)(r >> 16);
}
__device__ __forceinline__ float bf2f(unsigned short s) {
  return __uint_as_float(((unsigned int)s) << 16);
}
__device__ __forceinline__ float dsigmoid(float x) {
  return 1.0f / (1.0f + __expf(-x));
}

// ---------- generic f32 -> bf16 convert ----------
__global__ void cvt_f32_bf16_kernel(const float* __restrict__ in,
                                    unsigned short* __restrict__ out, long n) {
  long i = (long)blockIdx.x * blockDim.x + threadIdx.x;
  if (i < n) out[i] = f2bf(in[i]);
}

// ---------- f32 -> bf16 with column padding (row length N -> ldw) ----------
__global__ void cvt_pad_bf16_kernel(const float* __restrict__ in,
                                    unsigned short* __restrict__ out,
                                    int K, int N, int ldw) {
  long i = (long)blockIdx.x * blockDim.x + threadIdx.x;
  if (i >= (long)K * ldw) return;
  int k = (int)(i / ldw), n = (int)(i % ldw);
  out[i] = (n < N) ? f2bf(in[(size_t)k * N + n]) : (unsigned short)0;
}

// ---------- bias combine ----------
__global__ void bias_add_kernel(const float* __restrict__ a,
                                const float* __restrict__ b,
                                float* __restrict__ out, int n) {
  int i = blockIdx.x * blockDim.x + threadIdx.x;
  if (i < n) out[i] = a[i] + b[i];
}

// ---------- mean over P ----------
__global__ void mean_kernel(const float* __restrict__ features,
                            float* __restrict__ mean_f,
                            unsigned short* __restrict__ mean_b) {
  int idx = blockIdx.x * blockDim.x + threadIdx.x;   // B*D
  if (idx >= 128 * 2048) return;
  int b = idx >> 11, d = idx & 2047;
  const float* fp = features + (size_t)b * 196 * 2048 + d;
  float s = 0.f;
  for (int p = 0; p < 196; ++p) s += fp[(size_t)p * 2048];
  s *= (1.0f / 196.0f);
  mean_f[idx] = s;
  mean_b[idx] = f2bf(s);
}

// ---------- WMMA GEMM: out = act(A[MxK]bf16 @ W[Kxldw]bf16 + bias) * mask ----------
// Block tile 128x64, K-step 32, 8 waves (wave32): each wave owns a 32x32 tile
// (2x2 x v_wmma_f32_16x16x32_bf16). Double-buffered LDS, 1 barrier/iteration.
// ldw is a multiple of 64 so all W loads are unconditional b128.
#define TM 128
#define TN 64
#define TK 32
#define LDSS 40   // padded LDS row stride in shorts (80B rows, 16B aligned)

__global__ __launch_bounds__(256) void wmma_gemm_kernel(
    const unsigned short* __restrict__ A,
    const unsigned short* __restrict__ W,
    const float* __restrict__ bias,
    float* __restrict__ out, long out_stride,
    int M, int N, int K, int ldw,
    int act,                       // 0=none, 1=sigmoid
    const int* __restrict__ mask_len, int t) {
  __shared__ __align__(16) unsigned short ldsA[2][TM * LDSS];
  __shared__ __align__(16) unsigned short ldsW[2][TN * LDSS];

  const int tid   = threadIdx.x;
  const int lane  = tid & 31;
  const int wid   = tid >> 5;
  const int waveM = wid & 3;       // 4 M-bands of 32 rows
  const int waveN = wid >> 2;      // 2 N-bands of 32 cols
  const int blockN = blockIdx.x * TN;
  const int blockM = blockIdx.y * TM;

  v8f acc[2][2];
#pragma unroll
  for (int mi = 0; mi < 2; ++mi)
#pragma unroll
    for (int ni = 0; ni < 2; ++ni)
#pragma unroll
      for (int v = 0; v < 8; ++v) acc[mi][ni][v] = 0.0f;

  const int rl = lane & 15;
  const int kb = (lane >> 4) * 16;

  const int ar  = tid >> 1;            // A staging: row
  const int ac0 = (tid & 1) * 16;      // A staging: 16-short chunk
  const int wk  = tid >> 3;            // W staging: k row
  const int wn0 = (tid & 7) * 8;       // W staging: 8-col group

  // ---- prologue: stage K-tile 0 into buffer 0 (synchronous) ----
  {
    const unsigned short* gp = A + (size_t)(blockM + ar) * K + ac0;
    uint4 a0 = *reinterpret_cast<const uint4*>(gp);
    uint4 a1 = *reinterpret_cast<const uint4*>(gp + 8);
    *reinterpret_cast<uint4*>(&ldsA[0][ar * LDSS + ac0])     = a0;
    *reinterpret_cast<uint4*>(&ldsA[0][ar * LDSS + ac0 + 8]) = a1;
    const unsigned short* wgp = W + (size_t)wk * ldw + blockN + wn0;
    union { uint4 v; unsigned short s[8]; } wu;
    wu.v = *reinterpret_cast<const uint4*>(wgp);
#pragma unroll
    for (int j = 0; j < 8; ++j) ldsW[0][(wn0 + j) * LDSS + wk] = wu.s[j];
  }
  __syncthreads();

  const int nSteps = K / TK;
  for (int it = 0; it < nSteps; ++it) {
    const int cur = it & 1, nxt = cur ^ 1;
    const int kkn = (it + 1) * TK;
    const bool more = (it + 1) < nSteps;

    // ---- issue next-tile fetches before computing current tile ----
    uint4 pa0, pa1;
    union { uint4 v; unsigned short s[8]; } wu;
    if (more) {
#if defined(USE_ASYNC_LDS)
      // A tile: async DMA global->LDS (ASYNCcnt), overlaps with WMMA below
#pragma unroll
      for (int j = 0; j < 2; ++j) {
        int row   = wid * 16 + j * 8 + (lane >> 2);
        int chunk = lane & 3;
        const unsigned short* gsrc =
            A + (size_t)(blockM + row) * K + kkn + chunk * 8;
        unsigned short* ldst = &ldsA[nxt][row * LDSS + chunk * 8];
        __builtin_amdgcn_global_load_async_to_lds_b128(
            (as1_v4i)gsrc, (as3_v4i)ldst, 0, 0);
      }
#else
      const unsigned short* gp = A + (size_t)(blockM + ar) * K + kkn + ac0;
      pa0 = *reinterpret_cast<const uint4*>(gp);
      pa1 = *reinterpret_cast<const uint4*>(gp + 8);
#endif
      const unsigned short* wgp = W + (size_t)(kkn + wk) * ldw + blockN + wn0;
      wu.v = *reinterpret_cast<const uint4*>(wgp);
    }

    // ---- compute current tile ----
    v16bf afrag[2], bfrag[2];
#pragma unroll
    for (int mi = 0; mi < 2; ++mi) {
      int m = waveM * 32 + mi * 16 + rl;
      const uint4* ap =
          reinterpret_cast<const uint4*>(&ldsA[cur][m * LDSS + kb]);
      Frag128 fc; fc.q[0] = ap[0]; fc.q[1] = ap[1];
      afrag[mi] = fc.b;
    }
#pragma unroll
    for (int ni = 0; ni < 2; ++ni) {
      int n = waveN * 32 + ni * 16 + rl;
      const uint4* bp =
          reinterpret_cast<const uint4*>(&ldsW[cur][n * LDSS + kb]);
      Frag128 fc; fc.q[0] = bp[0]; fc.q[1] = bp[1];
      bfrag[ni] = fc.b;
    }
#pragma unroll
    for (int mi = 0; mi < 2; ++mi)
#pragma unroll
      for (int ni = 0; ni < 2; ++ni)
        acc[mi][ni] = __builtin_amdgcn_wmma_f32_16x16x32_bf16(
            false, afrag[mi], false, bfrag[ni], (short)0, acc[mi][ni],
            false, false);

    // ---- commit next tile to the other buffer ----
    if (more) {
#if !defined(USE_ASYNC_LDS)
      *reinterpret_cast<uint4*>(&ldsA[nxt][ar * LDSS + ac0])     = pa0;
      *reinterpret_cast<uint4*>(&ldsA[nxt][ar * LDSS + ac0 + 8]) = pa1;
#endif
#pragma unroll
      for (int j = 0; j < 8; ++j) ldsW[nxt][(wn0 + j) * LDSS + wk] = wu.s[j];
      wait_async0();
      __syncthreads();
    }
  }

  // epilogue: C layout -> lane = N, VGPR v = M (v | v+8 for upper half-wave)
  const int nl = lane & 15;
  const int mb = (lane >> 4) * 8;
#pragma unroll
  for (int mi = 0; mi < 2; ++mi)
#pragma unroll
    for (int ni = 0; ni < 2; ++ni) {
      int col = blockN + waveN * 32 + ni * 16 + nl;
      if (col < N) {
        float bv = bias ? bias[col] : 0.0f;
#pragma unroll
        for (int v = 0; v < 8; ++v) {
          int row = blockM + waveM * 32 + mi * 16 + mb + v;
          float val = acc[mi][ni][v] + bv;
          if (act == 1) val = dsigmoid(val);
          if (mask_len) val *= (t < mask_len[row] - 1) ? 1.0f : 0.0f;
          out[(size_t)row * out_stride + col] = val;
        }
      }
    }
}

// ---------- attention: e -> softmax -> alphas out -> awe ----------
__global__ __launch_bounds__(256) void attend_kernel(
    const float* __restrict__ att1,        // (B*P) x 512
    const float* __restrict__ att2,        // B x 512
    const float* __restrict__ att_w,       // 512
    const float* __restrict__ att_b,       // 1
    const unsigned short* __restrict__ featb, // B x P x 2048 bf16 (L2-resident)
    const int* __restrict__ lengths,
    float* __restrict__ awe,               // B x 2048
    float* __restrict__ alphas_out,        // (B,T,P) region of d_out
    int t) {
  __shared__ float s_e[196];
  __shared__ float s_red[256];
  const int b = blockIdx.x;
  const int tid = threadIdx.x;
  const int lane = tid & 31, wid = tid >> 5;

  // e[p] = sum_a relu(att1 + att2) * att_w  (one wave per p, lane-split over a)
  for (int p = wid; p < 196; p += 8) {
    const float* a1 = att1 + ((size_t)(b * 196 + p)) * 512;
    const float* a2 = att2 + (size_t)b * 512;
    float s = 0.f;
    for (int a = lane; a < 512; a += 32) {
      float v = a1[a] + a2[a];
      s += fmaxf(v, 0.0f) * att_w[a];
    }
#pragma unroll
    for (int off = 16; off > 0; off >>= 1) s += __shfl_down(s, off, 32);
    if (lane == 0) s_e[p] = s + att_b[0];
  }
  __syncthreads();

  // softmax over P=196
  float m = -1e30f;
  for (int p = tid; p < 196; p += 256) m = fmaxf(m, s_e[p]);
  s_red[tid] = m; __syncthreads();
  for (int o = 128; o > 0; o >>= 1) {
    if (tid < o) s_red[tid] = fmaxf(s_red[tid], s_red[tid + o]);
    __syncthreads();
  }
  m = s_red[0]; __syncthreads();
  float sum = 0.f;
  for (int p = tid; p < 196; p += 256) {
    float e = __expf(s_e[p] - m);
    s_e[p] = e; sum += e;
  }
  s_red[tid] = sum; __syncthreads();
  for (int o = 128; o > 0; o >>= 1) {
    if (tid < o) s_red[tid] += s_red[tid + o];
    __syncthreads();
  }
  float inv = 1.0f / s_red[0];
  float mf = (t < lengths[b] - 1) ? 1.0f : 0.0f;
  for (int p = tid; p < 196; p += 256) {
    float al = s_e[p] * inv;
    s_e[p] = al;
    alphas_out[((size_t)b * 19 + t) * 196 + p] = al * mf;
  }
  __syncthreads();

  // awe[b,d] = sum_p alpha[p] * feat_bf16[b,p,d]  (b128 loads, 8 ch/thread)
  const unsigned short* fb = featb + (size_t)b * 196 * 2048;
  const int d0 = tid * 8;
  float acc[8];
#pragma unroll
  for (int i = 0; i < 8; ++i) acc[i] = 0.f;
  for (int p = 0; p < 196; ++p) {
    float al = s_e[p];
    union { uint4 v; unsigned short s[8]; } u;
    u.v = *reinterpret_cast<const uint4*>(fb + (size_t)p * 2048 + d0);
#pragma unroll
    for (int i = 0; i < 8; ++i) acc[i] += al * bf2f(u.s[i]);
  }
#pragma unroll
  for (int i = 0; i < 8; ++i) awe[(size_t)b * 2048 + d0 + i] = acc[i];
}

// ---------- xin = [emb[cap[b,t]] | gate*awe | h] as bf16 ----------
__global__ void xin_build_kernel(const float* __restrict__ emb,
                                 const int* __restrict__ captions,
                                 const float* __restrict__ gate,
                                 const float* __restrict__ awe,
                                 const unsigned short* __restrict__ h_bf16,
                                 unsigned short* __restrict__ xin, int t) {
  int idx = blockIdx.x * blockDim.x + threadIdx.x;   // B*3072
  if (idx >= 128 * 3072) return;
  int b = idx / 3072, j = idx % 3072;
  unsigned short v;
  if (j < 512) {
    int tok = captions[b * 20 + t];
    v = f2bf(emb[(size_t)tok * 512 + j]);
  } else if (j < 2560) {
    int d = j - 512;
    v = f2bf(gate[(size_t)b * 2048 + d] * awe[(size_t)b * 2048 + d]);
  } else {
    v = h_bf16[b * 512 + (j - 2560)];
  }
  xin[(size_t)b * 3072 + j] = v;
}

// ---------- LSTM pointwise + masked commit ----------
__global__ void lstm_kernel(const float* __restrict__ gates,
                            float* __restrict__ h, float* __restrict__ c,
                            unsigned short* __restrict__ h_bf16,
                            unsigned short* __restrict__ hnew_bf16,
                            const int* __restrict__ lengths, int t) {
  int idx = blockIdx.x * blockDim.x + threadIdx.x;   // B*H
  if (idx >= 128 * 512) return;
  int b = idx >> 9, j = idx & 511;
  const float* g = gates + (size_t)b * 2048;
  float ig = dsigmoid(g[j]);
  float fg = dsigmoid(g[512 + j]);
  float gg = tanhf(g[1024 + j]);
  float og = dsigmoid(g[1536 + j]);
  float cn = fg * c[idx] + ig * gg;
  float hn = og * tanhf(cn);
  hnew_bf16[idx] = f2bf(hn);                 // y GEMM always uses h_new
  bool mk = t < lengths[b] - 1;
  float hc = mk ? hn : h[idx];
  float cc = mk ? cn : c[idx];
  h[idx] = hc; c[idx] = cc;
  h_bf16[idx] = f2bf(hc);                    // next-step GEMMs use committed h
}

// ---------- host ----------
static inline int cdiv(long a, long b) { return (int)((a + b - 1) / b); }

extern "C" void kernel_launch(void* const* d_in, const int* in_sizes, int n_in,
                              void* d_out, int out_size, void* d_ws, size_t ws_size,
                              hipStream_t stream) {
  const float* features = (const float*)d_in[0];
  const int*   captions = (const int*)d_in[1];
  const int*   lengths  = (const int*)d_in[2];
  const float* emb      = (const float*)d_in[3];
  const float* h_fc_w   = (const float*)d_in[4];
  const float* h_fc_b   = (const float*)d_in[5];
  const float* c_fc_w   = (const float*)d_in[6];
  const float* c_fc_b   = (const float*)d_in[7];
  const float* enc_w    = (const float*)d_in[8];
  const float* enc_b    = (const float*)d_in[9];
  const float* dec_w    = (const float*)d_in[10];
  const float* dec_b    = (const float*)d_in[11];
  const float* att_w    = (const float*)d_in[12];
  const float* att_b    = (const float*)d_in[13];
  const float* beta_w   = (const float*)d_in[14];
  const float* beta_b   = (const float*)d_in[15];
  const float* w_ih     = (const float*)d_in[16];
  const float* b_ih     = (const float*)d_in[17];
  const float* w_hh     = (const float*)d_in[18];
  const float* b_hh     = (const float*)d_in[19];
  const float* cls_w    = (const float*)d_in[20];
  const float* cls_b    = (const float*)d_in[21];

  const long B = 128, P = 196, D = 2048, Adim = 512, H = 512, V = 10000, T = 19;
  const long Vpad = 10048;                 // V rounded up to multiple of TN

  float* y_out      = (float*)d_out;                       // (B,T,V)
  float* alphas_out = (float*)d_out + B * T * V;           // (B,T,P)

  // workspace carve-out
  char* wp = (char*)d_ws;
  auto alloc = [&](size_t bytes) {
    char* p = wp; wp += (bytes + 255) & ~(size_t)255; return p;
  };
  unsigned short* featb  = (unsigned short*)alloc(B * P * D * 2);
  float*          att1   = (float*)alloc(B * P * Adim * 4);
  unsigned short* encb   = (unsigned short*)alloc(D * Adim * 2);
  unsigned short* hfcb   = (unsigned short*)alloc(D * H * 2);
  unsigned short* cfcb   = (unsigned short*)alloc(D * H * 2);
  unsigned short* decb   = (unsigned short*)alloc(H * Adim * 2);
  unsigned short* betab  = (unsigned short*)alloc(H * D * 2);
  unsigned short* wstkb  = (unsigned short*)alloc(3072 * 2048 * 2);
  unsigned short* clsb   = (unsigned short*)alloc(H * Vpad * 2);
  float*          bcomb  = (float*)alloc(2048 * 4);
  float*          meanf  = (float*)alloc(B * D * 4);
  unsigned short* meanb  = (unsigned short*)alloc(B * D * 2);
  float*          hbuf   = (float*)alloc(B * H * 4);
  float*          cbuf   = (float*)alloc(B * H * 4);
  unsigned short* hb16   = (unsigned short*)alloc(B * H * 2);
  unsigned short* hnb16  = (unsigned short*)alloc(B * H * 2);
  float*          att2   = (float*)alloc(B * Adim * 4);
  float*          awe    = (float*)alloc(B * D * 4);
  float*          gate   = (float*)alloc(B * D * 4);
  unsigned short* xinb   = (unsigned short*)alloc(B * 3072 * 2);
  float*          gates  = (float*)alloc(B * 2048 * 4);

  auto cvt = [&](const float* in, unsigned short* out, long n) {
    cvt_f32_bf16_kernel<<<cdiv(n, 256), 256, 0, stream>>>(in, out, n);
  };

  // one-time conversions (features -> bf16 makes the 19x-reused tensor L2-resident)
  cvt(features, featb, B * P * D);
  cvt(enc_w, encb, D * Adim);
  cvt(h_fc_w, hfcb, D * H);
  cvt(c_fc_w, cfcb, D * H);
  cvt(dec_w, decb, H * Adim);
  cvt(beta_w, betab, H * D);
  cvt(w_ih, wstkb, 2560 * 2048);                        // rows 0..2559
  cvt(w_hh, wstkb + (size_t)2560 * 2048, 512 * 2048);   // rows 2560..3071
  cvt_pad_bf16_kernel<<<cdiv(H * Vpad, 256), 256, 0, stream>>>(
      cls_w, clsb, (int)H, (int)V, (int)Vpad);          // pad to branch-free tiles
  bias_add_kernel<<<cdiv(2048, 256), 256, 0, stream>>>(b_ih, b_hh, bcomb, 2048);

  mean_kernel<<<cdiv(B * D, 256), 256, 0, stream>>>(features, meanf, meanb);

  auto gemm = [&](const unsigned short* Am, const unsigned short* Wm,
                  const float* bias, float* out, long ostride,
                  int M, int N, int K, int ldw, int act,
                  const int* mask, int t) {
    dim3 grid(cdiv(N, TN), M / TM);
    wmma_gemm_kernel<<<grid, 256, 0, stream>>>(Am, Wm, bias, out, ostride,
                                               M, N, K, ldw, act, mask, t);
  };

  // h0, c0 = mean_f @ {h,c}_fc_w + b
  gemm(meanb, hfcb, h_fc_b, hbuf, H, (int)B, (int)H, (int)D, (int)H, 0, nullptr, 0);
  gemm(meanb, cfcb, c_fc_b, cbuf, H, (int)B, (int)H, (int)D, (int)H, 0, nullptr, 0);
  cvt(hbuf, hb16, B * H);

  // att1 = features @ enc_w + enc_b   (M = B*P = 25088)
  gemm(featb, encb, enc_b, att1, Adim, (int)(B * P), (int)Adim, (int)D,
       (int)Adim, 0, nullptr, 0);

  for (int t = 0; t < (int)T; ++t) {
    // att2 = h @ dec_w + dec_b
    gemm(hb16, decb, dec_b, att2, Adim, (int)B, (int)Adim, (int)H, (int)Adim,
         0, nullptr, 0);
    // e/softmax/alphas/awe
    attend_kernel<<<(int)B, 256, 0, stream>>>(att1, att2, att_w, att_b, featb,
                                              lengths, awe, alphas_out, t);
    // gate = sigmoid(h @ beta_w + beta_b)
    gemm(hb16, betab, beta_b, gate, D, (int)B, (int)D, (int)H, (int)D,
         1, nullptr, 0);
    // xin = [emb | gate*awe | h]
    xin_build_kernel<<<cdiv(B * 3072, 256), 256, 0, stream>>>(
        emb, captions, gate, awe, hb16, xinb, t);
    // gates = xin @ [w_ih; w_hh] + (b_ih + b_hh)
    gemm(xinb, wstkb, bcomb, gates, 2048, (int)B, 2048, 3072, 2048,
         0, nullptr, 0);
    // LSTM pointwise + masked commit
    lstm_kernel<<<cdiv(B * H, 256), 256, 0, stream>>>(gates, hbuf, cbuf,
                                                      hb16, hnb16, lengths, t);
    // y[:, t, :] = (h_new @ cls_w + cls_b) * mask  (strided straight into d_out)
    gemm(hnb16, clsb, cls_b, y_out + (size_t)t * V, T * V,
         (int)B, (int)V, (int)H, (int)Vpad, 0, lengths, t);
  }
}
